// ProbSparseAttention_13125420057124
// MI455X (gfx1250) — compile-verified, matrix operand order checked by reading, not run on hardware
//
#include <hip/hip_runtime.h>
#include <hip/hip_bf16.h>

// ---------------------------------------------------------------------------
// ProbSparse attention for MI455X (gfx1250, wave32, WMMA).
//   - one-time fp32 -> bf16 conversion of x / Wq / Wk / Wv (and Q/K at the
//     projection epilogue), so GEMM hot loops have zero conversion VALU
//   - GEMM fragments loaded straight from global bf16 (2x 16B per lane);
//     192MB L2 serves all cross-block tile reuse (bf16 working set < 10MB)
//   - v_wmma_f32_16x16x32_bf16 everywhere it counts; softmax / context /
//     output projection stay fp32
// ---------------------------------------------------------------------------

typedef __attribute__((ext_vector_type(16))) __bf16 v16bf;
typedef __attribute__((ext_vector_type(8)))  __bf16 v8bf;
typedef __attribute__((ext_vector_type(8)))  float  v8f;

#define D_MODEL 512
#define N_HEADS 8
#define DK      64
#define SEQ_L   2048
#define BATCH   2
#define BH      (BATCH * N_HEADS)
#define NSAMP   10240
#define SCALE   0.125f            // 1/sqrt(64)
#define NEG_INF (-3.402823466e38f)

// ---- fp32 -> bf16 (round-to-nearest-even bias trick) ----------------------
__device__ __forceinline__ unsigned short bf16bits(float f) {
  union { float f; unsigned u; } v; v.f = f;
  unsigned r = v.u + 0x7FFFu + ((v.u >> 16) & 1u);
  return (unsigned short)(r >> 16);
}
__device__ __forceinline__ unsigned pk2(float a, float b) {
  return (unsigned)bf16bits(a) | ((unsigned)bf16bits(b) << 16);
}
__device__ __forceinline__ __bf16 f2bf(float f) {
  return __builtin_bit_cast(__bf16, bf16bits(f));
}

// WMMA 16-bit A/B fragment straight from global bf16 (row-major, stride ld):
// lane L: row = L%16, half = L/16; halves 0..7 = K[8h..8h+7],
//                                  halves 8..15 = K[16+8h..16+8h+7]
// => two aligned 16-byte global loads per lane.
__device__ __forceinline__ v16bf load_frag_g(const __bf16* __restrict__ g,
                                             int ld) {
  const int lane = threadIdx.x & 31;
  const int row  = lane & 15;
  const int half = lane >> 4;
  const __bf16* base = g + (size_t)row * ld + 8 * half;
  v8bf lo = *(const v8bf*)(base);
  v8bf hi = *(const v8bf*)(base + 16);
  return __builtin_shufflevector(lo, hi, 0, 1, 2, 3, 4, 5, 6, 7,
                                         8, 9, 10, 11, 12, 13, 14, 15);
}

__device__ __forceinline__ v8f wmma_bf16(v16bf a, v16bf b, v8f c) {
  return __builtin_amdgcn_wmma_f32_16x16x32_bf16(false, a, false, b,
                                                 (short)0, c, false, false);
}

// ---------------------------------------------------------------------------
// K0: one-time fp32 -> bf16 conversion (vectorized, n % 4 == 0)
// ---------------------------------------------------------------------------
__global__ __launch_bounds__(256) void cvt_kernel(
    const float* __restrict__ src, __bf16* __restrict__ dst, int n) {
  int i = (blockIdx.x * 256 + threadIdx.x) * 4;
  if (i < n) {
    float4 f = *(const float4*)(src + i);
    uint2 q; q.x = pk2(f.x, f.y); q.y = pk2(f.z, f.w);
    *(uint2*)(dst + i) = q;
  }
}

// ---------------------------------------------------------------------------
// K1: projections.  C[4096,512] = Xbf @ Wbf^T (+bias), scattered to per-head
// [b][h][l][d] as fp32 (+ bf16 copy for Q and K).  One wave per 16x64 strip,
// A fragment reused across 4 N-subtiles.  grid = (256, 8, 3).
// ---------------------------------------------------------------------------
__global__ __launch_bounds__(32) void proj_kernel(
    const __bf16* __restrict__ Xbf, const __bf16* __restrict__ Wbf,
    const float* __restrict__ bq, const float* __restrict__ bk,
    const float* __restrict__ bv,
    float* __restrict__ Qf, float* __restrict__ Kf, float* __restrict__ Vf,
    __bf16* __restrict__ Qbf, __bf16* __restrict__ Kbf) {
  const int mt    = blockIdx.x;            // 16-row tile over B*L
  const int nb    = blockIdx.y;            // 64-col block
  const int which = blockIdx.z;            // 0=Q 1=K 2=V
  const __bf16* W   = Wbf + (size_t)which * D_MODEL * D_MODEL;
  const float* bias = (which == 0) ? bq : (which == 1) ? bk : bv;
  float*      Out   = (which == 0) ? Qf : (which == 1) ? Kf : Vf;
  __bf16*     Obf   = (which == 0) ? Qbf : (which == 1) ? Kbf : nullptr;

  const int m0 = mt * 16;
  const int n0 = nb * 64;

  const v8f vzero = {0.f, 0.f, 0.f, 0.f, 0.f, 0.f, 0.f, 0.f};
  v8f acc[4];
#pragma unroll
  for (int t = 0; t < 4; ++t) acc[t] = vzero;

  for (int kb = 0; kb < D_MODEL; kb += 32) {
    v16bf a = load_frag_g(Xbf + (size_t)m0 * D_MODEL + kb, D_MODEL);
#pragma unroll
    for (int t = 0; t < 4; ++t) {
      v16bf b = load_frag_g(W + (size_t)(n0 + t * 16) * D_MODEL + kb, D_MODEL);
      acc[t] = wmma_bf16(a, b, acc[t]);
    }
  }

  // C layout: VGPR r -> row r + 8*(lane/16); lane%16 -> column.
  const int lane  = threadIdx.x & 31;
  const int nloc  = lane & 15;
  const int mhalf = (lane >> 4) * 8;
#pragma unroll
  for (int t = 0; t < 4; ++t) {
    const int e  = n0 + t * 16 + nloc;
    const float be = bias[e];
    const int h = e >> 6, d = e & 63;
#pragma unroll
    for (int r = 0; r < 8; ++r) {
      const int gm = m0 + mhalf + r;       // row in [0, B*L)
      const int bb = gm >> 11, l = gm & (SEQ_L - 1);
      const size_t idx = (((size_t)(bb * N_HEADS + h) * SEQ_L) + l) * DK + d;
      const float v = acc[t][r] + be;
      Out[idx] = v;
      if (Obf) Obf[idx] = f2bf(v);
    }
  }
}

// ---------------------------------------------------------------------------
// K2a: histogram of sampled indices.
// ---------------------------------------------------------------------------
__global__ __launch_bounds__(256) void count_kernel(
    const int* __restrict__ samp, int n, int* __restrict__ count) {
  int i = blockIdx.x * 256 + threadIdx.x;
  if (i < n) atomicAdd(&count[samp[i]], 1);
}

// K2b: kbar[bh,d] = (1/S) * sum_j count[j] * K[bh,j,d]
__global__ __launch_bounds__(64) void kbar_kernel(
    const float* __restrict__ Kf, const int* __restrict__ count,
    float* __restrict__ kbar) {
  const int bh = blockIdx.x, d = threadIdx.x;
  float acc = 0.f;
  for (int j = 0; j < SEQ_L; ++j) {
    int c = count[j];
    if (c) acc += (float)c * Kf[((size_t)bh * SEQ_L + j) * DK + d];
  }
  kbar[bh * DK + d] = acc * (1.0f / (float)NSAMP);
}

// ---------------------------------------------------------------------------
// K2c: M[l] = scale*(max over sampled j of q_l.k_j  -  q_l.kbar)
// One wave per 16 query rows per (b,h); full K sweep, masked column row-max.
// grid = (128, 16).
// ---------------------------------------------------------------------------
__global__ __launch_bounds__(32) void score_kernel(
    const __bf16* __restrict__ Qbf, const __bf16* __restrict__ Kbf,
    const float* __restrict__ Qf, const int* __restrict__ count,
    const float* __restrict__ kbar, float* __restrict__ Mbuf) {
  const int lt = blockIdx.x;               // 16-row query tile
  const int bh = blockIdx.y;
  const __bf16* Qh = Qbf + (size_t)bh * SEQ_L * DK;
  const __bf16* Kh = Kbf + (size_t)bh * SEQ_L * DK;
  const int lane = threadIdx.x & 31;

  v16bf aq0 = load_frag_g(Qh + (size_t)lt * 16 * DK + 0,  DK);
  v16bf aq1 = load_frag_g(Qh + (size_t)lt * 16 * DK + 32, DK);

  float runmax[8];
#pragma unroll
  for (int r = 0; r < 8; ++r) runmax[r] = NEG_INF;

  const v8f vzero = {0.f, 0.f, 0.f, 0.f, 0.f, 0.f, 0.f, 0.f};
  const int ncol = lane & 15;

  for (int jt = 0; jt < SEQ_L / 16; ++jt) {
    if (jt + 2 < SEQ_L / 16)               // cover streaming latency
      __builtin_prefetch(Kh + (size_t)(jt + 2) * 16 * DK, 0, 1);
    v8f c = vzero;
    v16bf b0 = load_frag_g(Kh + (size_t)jt * 16 * DK + 0,  DK);
    c = wmma_bf16(aq0, b0, c);
    v16bf b1 = load_frag_g(Kh + (size_t)jt * 16 * DK + 32, DK);
    c = wmma_bf16(aq1, b1, c);
    const bool sampled = count[jt * 16 + ncol] != 0;
    if (sampled) {
#pragma unroll
      for (int r = 0; r < 8; ++r) runmax[r] = fmaxf(runmax[r], c[r]);
    }
  }
  // max across the 16 lanes of each half-wave (columns of each row group)
#pragma unroll
  for (int off = 1; off < 16; off <<= 1) {
#pragma unroll
    for (int r = 0; r < 8; ++r)
      runmax[r] = fmaxf(runmax[r], __shfl_xor(runmax[r], off, 16));
  }
  if ((lane & 15) == 0) {
    const int mbase = (lane >> 4) * 8;
    const float* kb = kbar + bh * DK;
#pragma unroll 1
    for (int r = 0; r < 8; ++r) {
      const int l = lt * 16 + mbase + r;
      const float* qrow = Qf + ((size_t)bh * SEQ_L + l) * DK;
      float dot = 0.f;
      for (int d = 0; d < DK; ++d) dot += qrow[d] * kb[d];
      Mbuf[bh * SEQ_L + l] = SCALE * (runmax[r] - dot);
    }
  }
}

// ---------------------------------------------------------------------------
// K3: U[bh] = argmax_l M[bh,l]   (first index on ties, like jnp.argmax)
// ---------------------------------------------------------------------------
__global__ __launch_bounds__(256) void argmax_kernel(
    const float* __restrict__ Mbuf, int* __restrict__ U) {
  __shared__ float sv[256];
  __shared__ int   si[256];
  const int bh = blockIdx.x, t = threadIdx.x;
  float best = NEG_INF; int bidx = 0;
  for (int l = t; l < SEQ_L; l += 256) {
    float v = Mbuf[bh * SEQ_L + l];
    if (v > best) { best = v; bidx = l; }
  }
  sv[t] = best; si[t] = bidx;
  __syncthreads();
  for (int s = 128; s > 0; s >>= 1) {
    if (t < s) {
      if (sv[t + s] > sv[t] || (sv[t + s] == sv[t] && si[t + s] < si[t])) {
        sv[t] = sv[t + s]; si[t] = si[t + s];
      }
    }
    __syncthreads();
  }
  if (t == 0) U[bh] = si[0];
}

// ---------------------------------------------------------------------------
// K4: single-query attention per (b,h): softmax(q_U . K^T) @ V -> ctx[bh,64]
// ---------------------------------------------------------------------------
__global__ __launch_bounds__(256) void attn_kernel(
    const float* __restrict__ Qf, const float* __restrict__ Kf,
    const float* __restrict__ Vf, const int* __restrict__ U,
    float* __restrict__ ctx) {
  __shared__ float sbuf[SEQ_L];
  __shared__ float red[256];
  __shared__ float csh[DK];
  const int bh = blockIdx.x, t = threadIdx.x;
  const float* qr = Qf + ((size_t)bh * SEQ_L + U[bh]) * DK;

  float lmax = NEG_INF;
  for (int l = t; l < SEQ_L; l += 256) {
    const float* krow = Kf + ((size_t)bh * SEQ_L + l) * DK;
    float s = 0.f;
    for (int d = 0; d < DK; ++d) s += qr[d] * krow[d];
    s *= SCALE;
    sbuf[l] = s;
    lmax = fmaxf(lmax, s);
  }
  red[t] = lmax; __syncthreads();
  for (int s = 128; s > 0; s >>= 1) {
    if (t < s) red[t] = fmaxf(red[t], red[t + s]);
    __syncthreads();
  }
  const float gmax = red[0]; __syncthreads();

  float lsum = 0.f;
  for (int l = t; l < SEQ_L; l += 256) {
    float p = __expf(sbuf[l] - gmax);
    sbuf[l] = p;
    lsum += p;
  }
  red[t] = lsum; __syncthreads();
  for (int s = 128; s > 0; s >>= 1) {
    if (t < s) red[t] += red[t + s];
    __syncthreads();
  }
  const float inv = 1.0f / red[0];
  if (t < DK) csh[t] = 0.f;
  __syncthreads();
  for (int l = t; l < SEQ_L; l += 256) {
    const float p = sbuf[l] * inv;
    const float* vrow = Vf + ((size_t)bh * SEQ_L + l) * DK;
    for (int d = 0; d < DK; ++d) atomicAdd(&csh[d], p * vrow[d]);
  }
  __syncthreads();
  if (t < DK) ctx[bh * DK + t] = csh[t];
}

// ---------------------------------------------------------------------------
// K5: out[b,l,e] = ctx[b] . Wo[e,:] + bo[e]  (identical for all l)
// ---------------------------------------------------------------------------
__global__ __launch_bounds__(128) void out_kernel(
    const float* __restrict__ ctx, const float* __restrict__ Wo,
    const float* __restrict__ bo, float* __restrict__ out) {
  __shared__ float ytile[128];
  const int b = blockIdx.x, eb = blockIdx.y, t = threadIdx.x;
  const int e = eb * 128 + t;
  const float* c = ctx + (size_t)b * D_MODEL;   // [h*64+d] layout == e layout
  const float* w = Wo + (size_t)e * D_MODEL;
  float acc = bo[e];
  for (int d = 0; d < D_MODEL; ++d) acc += c[d] * w[d];
  ytile[t] = acc;
  __syncthreads();
  float y = ytile[t];
  for (int l = 0; l < SEQ_L; ++l)
    out[((size_t)(b * SEQ_L + l)) * D_MODEL + eb * 128 + t] = y;
}

// ---------------------------------------------------------------------------
extern "C" void kernel_launch(void* const* d_in, const int* in_sizes, int n_in,
                              void* d_out, int out_size, void* d_ws,
                              size_t ws_size, hipStream_t stream) {
  const float* x    = (const float*)d_in[0];
  const int*   samp = (const int*)d_in[1];
  const float* Wq = (const float*)d_in[2];
  const float* bq = (const float*)d_in[3];
  const float* Wk = (const float*)d_in[4];
  const float* bk = (const float*)d_in[5];
  const float* Wv = (const float*)d_in[6];
  const float* bv = (const float*)d_in[7];
  const float* Wo = (const float*)d_in[8];
  const float* bo = (const float*)d_in[9];
  float* out = (float*)d_out;
  const int nsamp = in_sizes[1];

  // --- workspace carve-up (256B-aligned slabs) -----------------------------
  char* p = (char*)d_ws;
  auto alloc = [&](size_t bytes) -> char* {
    char* r = p;
    p += (bytes + 255) & ~(size_t)255;
    return r;
  };
  const size_t QKV = (size_t)BH * SEQ_L * DK;      // 2,097,152 elements
  const size_t XN  = (size_t)BATCH * SEQ_L * D_MODEL;
  const size_t WN  = (size_t)D_MODEL * D_MODEL;

  float*  Qf   = (float*)alloc(QKV * 4);
  float*  Kf   = (float*)alloc(QKV * 4);
  float*  Vf   = (float*)alloc(QKV * 4);
  float*  Mbuf = (float*)alloc((size_t)BH * SEQ_L * 4);
  float*  kbar = (float*)alloc((size_t)BH * DK * 4);
  float*  ctx  = (float*)alloc((size_t)BH * DK * 4);
  int*    cnt  = (int*)alloc(SEQ_L * 4);
  int*    U    = (int*)alloc(BH * 4);
  __bf16* Xbf  = (__bf16*)alloc(XN * 2);
  __bf16* Wbf  = (__bf16*)alloc(3 * WN * 2);       // [Wq|Wk|Wv]
  __bf16* Qbf  = (__bf16*)alloc(QKV * 2);
  __bf16* Kbf  = (__bf16*)alloc(QKV * 2);

  hipMemsetAsync(cnt, 0, SEQ_L * sizeof(int), stream);

  // one-time bf16 conversions
  cvt_kernel<<<(int)(XN / 4 + 255) / 256, 256, 0, stream>>>(x, Xbf, (int)XN);
  cvt_kernel<<<(int)(WN / 4 + 255) / 256, 256, 0, stream>>>(Wq, Wbf, (int)WN);
  cvt_kernel<<<(int)(WN / 4 + 255) / 256, 256, 0, stream>>>(Wk, Wbf + WN, (int)WN);
  cvt_kernel<<<(int)(WN / 4 + 255) / 256, 256, 0, stream>>>(Wv, Wbf + 2 * WN, (int)WN);

  proj_kernel<<<dim3(BATCH * SEQ_L / 16, D_MODEL / 64, 3), 32, 0, stream>>>(
      Xbf, Wbf, bq, bk, bv, Qf, Kf, Vf, Qbf, Kbf);
  count_kernel<<<(nsamp + 255) / 256, 256, 0, stream>>>(samp, nsamp, cnt);
  kbar_kernel<<<BH, DK, 0, stream>>>(Kf, cnt, kbar);
  score_kernel<<<dim3(SEQ_L / 16, BH), 32, 0, stream>>>(Qbf, Kbf, Qf, cnt,
                                                        kbar, Mbuf);
  argmax_kernel<<<BH, 256, 0, stream>>>(Mbuf, U);
  attn_kernel<<<BH, 256, 0, stream>>>(Qf, Kf, Vf, U, ctx);
  out_kernel<<<dim3(BATCH, D_MODEL / 128), 128, 0, stream>>>(ctx, Wo, bo, out);
}
